// GraphConvolution_36584531428116
// MI455X (gfx1250) — matrix operand order repaired and checked by reading, not compiled, not run on hardware
//
#include <hip/hip_runtime.h>
#include <hip/hip_bf16.h>
#include <stdint.h>

// ---------------------------------------------------------------------------
// GraphConvolution node-layer forward for MI455X (gfx1250, wave32, WMMA).
//
//   out = ((0.5*(M1'+1)) (.) adj_v) @ (H_v@weight) + bias,
//   M1' = T*diag(gate)*T^T with diagonal forced to 1.  adj_e is unused.
//
// bf16 WMMA (v_wmma_f32_16x16x32_bf16), f32 accumulation.  T_bf16 (128 MB)
// fits the 192 MB L2, so the rank-8192 product streams from L2.
// Staging uses async global->LDS (ASYNCcnt) with double-buffered LDS so the
// copy of k-step t+1 overlaps the WMMAs of k-step t.  All LDS staging stores
// are packed 128-bit (ds_store_b128), except hw_kernel's transposed B tile (16-bit stores).
//
// Workspace layout (requires ws_size >= ~266 MB):
//   [0, 32KB)              gate            f32 [8192]
//   [1MB, 1MB+8MB)         HwT_bf16        bf16 [512][8192]   (H_v@weight)^T
//   [+ , +128MB)           T_bf16          bf16 [8192][8192]
//   [+ , +128MB)           M_bf16          bf16 [8192][8192]  fused A-matrix
// ---------------------------------------------------------------------------

#define NV   8192
#define NE   8192
#define DIN  512
#define DOUT 512

typedef __attribute__((ext_vector_type(16))) __bf16 v16bf;
typedef __attribute__((ext_vector_type(8)))  float  v8f;

// -------- async global->LDS support (guarded; falls back to uint4 copy) ----
#if defined(__HIP_DEVICE_COMPILE__)
#if defined(__has_builtin)
#if __has_builtin(__builtin_amdgcn_global_load_async_to_lds_b128)
#define ASYNC_LDS 1
#endif
#endif
#endif

// Parameter types per hipcc diagnostic: v4i (vector_size 16) pointers,
// global (AS1) source and LDS (AS3) destination.
typedef int v4i __attribute__((vector_size(16)));
typedef __attribute__((address_space(1))) v4i as1_v4i;
typedef __attribute__((address_space(3))) v4i as3_v4i;

static __device__ __forceinline__ void async_cp16(const void* g, void* l) {
#ifdef ASYNC_LDS
  __builtin_amdgcn_global_load_async_to_lds_b128((as1_v4i*)g, (as3_v4i*)l, 0, 0);
#else
  *(uint4*)l = *(const uint4*)g;
#endif
}
static __device__ __forceinline__ void wait_async() {
#ifdef ASYNC_LDS
#if __has_builtin(__builtin_amdgcn_s_wait_asynccnt)
  __builtin_amdgcn_s_wait_asynccnt(0);
#else
  asm volatile("s_wait_asynccnt 0" ::: "memory");
#endif
#endif
}

// -------- scalar conversions ------------------------------------------------
static __device__ __forceinline__ unsigned short f2bf(float f) {
  uint32_t u = __float_as_uint(f);
  u += 0x7FFFu + ((u >> 16) & 1u);          // round-to-nearest-even
  return (unsigned short)(u >> 16);
}
static __device__ __forceinline__ float bf2f(unsigned short h) {
  return __uint_as_float(((uint32_t)h) << 16);
}
static __device__ __forceinline__ uint32_t pack2(float lo, float hi) {
  return (uint32_t)f2bf(lo) | ((uint32_t)f2bf(hi) << 16);
}

// --- WMMA fragment loaders from LDS (ISA 7.12.2, 16-bit, wave32) -----------
// A 16x32: lane half 0 -> K 0-7 / 16-23, half 1 -> K 8-15 / 24-31.
static __device__ __forceinline__ v16bf load_frag_a(const unsigned short* lds,
                                                    int row, int half, int ld) {
  union { v16bf v; uint32_t u[8]; } f;
  const unsigned short* p = lds + row * ld;
#pragma unroll
  for (int pr = 0; pr < 8; ++pr) {
    int k2 = ((pr >> 2) << 4) + (half << 3) + ((pr & 3) << 1);
    f.u[pr] = *(const uint32_t*)(p + k2);
  }
  return f.v;
}
// B 32x16: lane half 0 -> K 0-7 / 16-23, half 1 -> K 8-15 / 24-31.
static __device__ __forceinline__ v16bf load_frag_b(const unsigned short* lds,
                                                    int row, int half, int ld) {
  union { v16bf v; uint32_t u[8]; } f;
  const unsigned short* p = lds + row * ld;
#pragma unroll
  for (int pr = 0; pr < 8; ++pr) {
    int k2 = ((pr >> 2) << 4) + (half << 3) + ((pr & 3) << 1);
    f.u[pr] = *(const uint32_t*)(p + k2);
  }
  return f.v;
}

// --- K0: T (f32) -> T_bf16 --------------------------------------------------
__global__ __launch_bounds__(256) void cvt_T_kernel(const float* __restrict__ T,
                                                    unsigned short* __restrict__ Tbf) {
  size_t base = ((size_t)blockIdx.x * 256 + threadIdx.x) * 8;
  float4 a = *(const float4*)(T + base);
  float4 b = *(const float4*)(T + base + 4);
  uint4 o;
  o.x = pack2(a.x, a.y); o.y = pack2(a.z, a.w);
  o.z = pack2(b.x, b.y); o.w = pack2(b.z, b.w);
  *(uint4*)(Tbf + base) = o;
}

// --- K1: gate[e] = dot(H_e[e,:], p) ----------------------------------------
__global__ __launch_bounds__(256) void gate_kernel(const float* __restrict__ He,
                                                   const float* __restrict__ p,
                                                   float* __restrict__ gate) {
  int e    = blockIdx.x * 8 + (threadIdx.x >> 5);
  int lane = threadIdx.x & 31;
  float s = 0.f;
#pragma unroll
  for (int j = 0; j < DIN / 32; ++j) {
    int idx = lane + 32 * j;
    s += He[(size_t)e * DIN + idx] * p[idx];
  }
#pragma unroll
  for (int off = 16; off > 0; off >>= 1) s += __shfl_down(s, off, 32);
  if (lane == 0) gate[e] = s;
}

// --- K2: HwT[n][m] = (H_v @ weight)^T, bf16, via WMMA (small: single-buffer)
__global__ __launch_bounds__(256) void hw_kernel(const float* __restrict__ Hv,
                                                 const float* __restrict__ W,
                                                 unsigned short* __restrict__ HwT) {
  constexpr int LD = 40;
  __shared__ unsigned short lA[128 * LD];
  __shared__ unsigned short lB[128 * LD];
  const int tid = threadIdx.x, lane = tid & 31, w = tid >> 5;
  const int half = (lane >> 4) & 1, mn = lane & 15;
  const int wr = w >> 2, wc = w & 3;
  const int m0 = blockIdx.x * 128, n0 = blockIdx.y * 128;

  v8f acc[4][2];
  for (int a = 0; a < 4; ++a) for (int b = 0; b < 2; ++b)
    for (int q = 0; q < 8; ++q) acc[a][b][q] = 0.f;

  const int srow = tid >> 1, skc = (tid & 1) << 4;
  const int bkk = tid >> 3, bnn = (tid & 7) << 4;

  for (int k0 = 0; k0 < DIN; k0 += 32) {
    __syncthreads();
    {   // A: H_v rows m0..m0+127, cols k0..k0+31, f32 -> bf16, b128 stores
      const float* src = Hv + (size_t)(m0 + srow) * DIN + k0 + skc;
      float4 a0 = *(const float4*)(src);
      float4 a1 = *(const float4*)(src + 4);
      float4 a2 = *(const float4*)(src + 8);
      float4 a3 = *(const float4*)(src + 12);
      uint4 o0, o1;
      o0.x = pack2(a0.x, a0.y); o0.y = pack2(a0.z, a0.w);
      o0.z = pack2(a1.x, a1.y); o0.w = pack2(a1.z, a1.w);
      o1.x = pack2(a2.x, a2.y); o1.y = pack2(a2.z, a2.w);
      o1.z = pack2(a3.x, a3.y); o1.w = pack2(a3.z, a3.w);
      unsigned short* dst = lA + srow * LD + skc;
      *(uint4*)(dst)     = o0;
      *(uint4*)(dst + 8) = o1;
    }
    {   // B: weight[k][n] -> lB[n][k] (transpose into LDS)
      const float* src = W + (size_t)(k0 + bkk) * DOUT + n0 + bnn;
#pragma unroll
      for (int j = 0; j < 16; ++j) lB[(bnn + j) * LD + bkk] = f2bf(src[j]);
    }
    __syncthreads();

    v16bf bf[2];
#pragma unroll
    for (int nt = 0; nt < 2; ++nt)
      bf[nt] = load_frag_b(lB, wc * 32 + nt * 16 + mn, half, LD);
#pragma unroll
    for (int mt = 0; mt < 4; ++mt) {
      v16bf af = load_frag_a(lA, wr * 64 + mt * 16 + mn, half, LD);
#pragma unroll
      for (int nt = 0; nt < 2; ++nt)
        acc[mt][nt] = __builtin_amdgcn_wmma_f32_16x16x32_bf16(
            false, af, false, bf[nt], (short)0, acc[mt][nt], false, false);
    }
  }
#pragma unroll
  for (int mt = 0; mt < 4; ++mt)
#pragma unroll
    for (int nt = 0; nt < 2; ++nt)
#pragma unroll
      for (int r = 0; r < 8; ++r) {
        int m = m0 + wr * 64 + mt * 16 + r + (half << 3);
        int n = n0 + wc * 32 + nt * 16 + mn;
        HwT[(size_t)n * NV + m] = f2bf(acc[mt][nt][r]);
      }
}

// --- K3 (pass A): M = bf16(0.5*(fixdiag(Tg @ T^T)+1) .* adj_v) --------------
// Double-buffered LDS; B operand async; gated A staged with b128 stores.
__global__ __launch_bounds__(256) void passA_kernel(const unsigned short* __restrict__ Tbf,
                                                    const float* __restrict__ gate,
                                                    const float* __restrict__ adj_v,
                                                    unsigned short* __restrict__ Mout) {
  constexpr int LD = 40;
  __shared__ unsigned short lA[2][128 * LD];
  __shared__ unsigned short lB[2][128 * LD];
  const int tid = threadIdx.x, lane = tid & 31, w = tid >> 5;
  const int half = (lane >> 4) & 1, mn = lane & 15;
  const int wr = w >> 2, wc = w & 3;
  const int i0 = blockIdx.x * 128, j0 = blockIdx.y * 128;

  v8f acc[4][2];
  for (int a = 0; a < 4; ++a) for (int b = 0; b < 2; ++b)
    for (int q = 0; q < 8; ++q) acc[a][b][q] = 0.f;

  const int srow = tid >> 1, skc = (tid & 1) << 4;

  auto stage = [&](int buf, int k0) {
    {   // A: bf16(T[i][k] * gate[k]); packed pairs, two ds_store_b128
      const unsigned short* src = Tbf + (size_t)(i0 + srow) * NE + k0 + skc;
      unsigned short tmp[16];
      *(uint4*)(tmp)     = *(const uint4*)(src);
      *(uint4*)(tmp + 8) = *(const uint4*)(src + 8);
      const float* gp = gate + k0 + skc;
      float4 g0 = *(const float4*)(gp);
      float4 g1 = *(const float4*)(gp + 4);
      float4 g2 = *(const float4*)(gp + 8);
      float4 g3 = *(const float4*)(gp + 12);
      uint4 o0, o1;
      o0.x = pack2(bf2f(tmp[0]) * g0.x,  bf2f(tmp[1]) * g0.y);
      o0.y = pack2(bf2f(tmp[2]) * g0.z,  bf2f(tmp[3]) * g0.w);
      o0.z = pack2(bf2f(tmp[4]) * g1.x,  bf2f(tmp[5]) * g1.y);
      o0.w = pack2(bf2f(tmp[6]) * g1.z,  bf2f(tmp[7]) * g1.w);
      o1.x = pack2(bf2f(tmp[8]) * g2.x,  bf2f(tmp[9]) * g2.y);
      o1.y = pack2(bf2f(tmp[10]) * g2.z, bf2f(tmp[11]) * g2.w);
      o1.z = pack2(bf2f(tmp[12]) * g3.x, bf2f(tmp[13]) * g3.y);
      o1.w = pack2(bf2f(tmp[14]) * g3.z, bf2f(tmp[15]) * g3.w);
      unsigned short* dst = &lA[buf][srow * LD + skc];
      *(uint4*)(dst)     = o0;
      *(uint4*)(dst + 8) = o1;
    }
    {   // B: T^T tile == row-major T rows j0..j0+127 (async copy)
      const unsigned short* src = Tbf + (size_t)(j0 + srow) * NE + k0 + skc;
      unsigned short* dst = &lB[buf][srow * LD + skc];
      async_cp16(src, dst);
      async_cp16(src + 8, dst + 8);
    }
  };

  stage(0, 0);
  for (int it = 0; it < NE / 32; ++it) {
    const int cur = it & 1;
    wait_async();
    __syncthreads();
    if (it + 1 < NE / 32) stage(cur ^ 1, (it + 1) * 32);

    v16bf bf[2];
#pragma unroll
    for (int nt = 0; nt < 2; ++nt)
      bf[nt] = load_frag_b(lB[cur], wc * 32 + nt * 16 + mn, half, LD);
#pragma unroll
    for (int mt = 0; mt < 4; ++mt) {
      v16bf af = load_frag_a(lA[cur], wr * 64 + mt * 16 + mn, half, LD);
#pragma unroll
      for (int nt = 0; nt < 2; ++nt)
        acc[mt][nt] = __builtin_amdgcn_wmma_f32_16x16x32_bf16(
            false, af, false, bf[nt], (short)0, acc[mt][nt], false, false);
    }
  }
  // epilogue: diag->1, 0.5*(x+1)*adj_v, store bf16
#pragma unroll
  for (int mt = 0; mt < 4; ++mt)
#pragma unroll
    for (int nt = 0; nt < 2; ++nt)
#pragma unroll
      for (int r = 0; r < 8; ++r) {
        int i = i0 + wr * 64 + mt * 16 + r + (half << 3);
        int j = j0 + wc * 32 + nt * 16 + mn;
        float v = acc[mt][nt][r];
        if (i == j) v = 1.0f;
        v = 0.5f * (v + 1.0f) * adj_v[(size_t)i * NV + j];
        Mout[(size_t)i * NV + j] = f2bf(v);
      }
}

// --- K4 (pass B): out = M @ Hw + bias (f32 out) -----------------------------
// Double-buffered LDS; both operands staged via async global->LDS.
__global__ __launch_bounds__(256) void passB_kernel(const unsigned short* __restrict__ Mbf,
                                                    const unsigned short* __restrict__ HwT,
                                                    const float* __restrict__ bias,
                                                    float* __restrict__ out) {
  constexpr int LD = 40;
  __shared__ unsigned short lA[2][128 * LD];
  __shared__ unsigned short lB[2][128 * LD];
  const int tid = threadIdx.x, lane = tid & 31, w = tid >> 5;
  const int half = (lane >> 4) & 1, mn = lane & 15;
  const int wr = w >> 2, wc = w & 3;
  const int i0 = blockIdx.x * 128, n0 = blockIdx.y * 128;

  v8f acc[4][2];
  for (int a = 0; a < 4; ++a) for (int b = 0; b < 2; ++b)
    for (int q = 0; q < 8; ++q) acc[a][b][q] = 0.f;

  const int srow = tid >> 1, skc = (tid & 1) << 4;

  auto stage = [&](int buf, int k0) {
    const unsigned short* sa = Mbf + (size_t)(i0 + srow) * NV + k0 + skc;
    unsigned short* da = &lA[buf][srow * LD + skc];
    async_cp16(sa, da);
    async_cp16(sa + 8, da + 8);
    const unsigned short* sb = HwT + (size_t)(n0 + srow) * NV + k0 + skc;
    unsigned short* db = &lB[buf][srow * LD + skc];
    async_cp16(sb, db);
    async_cp16(sb + 8, db + 8);
  };

  stage(0, 0);
  for (int it = 0; it < NV / 32; ++it) {
    const int cur = it & 1;
    wait_async();
    __syncthreads();
    if (it + 1 < NV / 32) stage(cur ^ 1, (it + 1) * 32);

    v16bf bf[2];
#pragma unroll
    for (int nt = 0; nt < 2; ++nt)
      bf[nt] = load_frag_b(lB[cur], wc * 32 + nt * 16 + mn, half, LD);
#pragma unroll
    for (int mt = 0; mt < 4; ++mt) {
      v16bf af = load_frag_a(lA[cur], wr * 64 + mt * 16 + mn, half, LD);
#pragma unroll
      for (int nt = 0; nt < 2; ++nt)
        acc[mt][nt] = __builtin_amdgcn_wmma_f32_16x16x32_bf16(
            false, af, false, bf[nt], (short)0, acc[mt][nt], false, false);
    }
  }
#pragma unroll
  for (int mt = 0; mt < 4; ++mt)
#pragma unroll
    for (int nt = 0; nt < 2; ++nt)
#pragma unroll
      for (int r = 0; r < 8; ++r) {
        int i = i0 + wr * 64 + mt * 16 + r + (half << 3);
        int n = n0 + wc * 32 + nt * 16 + mn;
        out[(size_t)i * DOUT + n] = acc[mt][nt][r] + bias[n];
      }
}

// ---------------------------------------------------------------------------
extern "C" void kernel_launch(void* const* d_in, const int* in_sizes, int n_in,
                              void* d_out, int out_size, void* d_ws, size_t ws_size,
                              hipStream_t stream) {
  const float* H_v    = (const float*)d_in[0];
  const float* H_e    = (const float*)d_in[1];
  /* d_in[2] = adj_e: unused by node-layer path */
  const float* adj_v  = (const float*)d_in[3];
  const float* T      = (const float*)d_in[4];
  const float* weight = (const float*)d_in[5];
  const float* p      = (const float*)d_in[6];
  const float* bias   = (const float*)d_in[7];
  float* out = (float*)d_out;

  char* ws = (char*)d_ws;
  float*          gate = (float*)(ws);
  unsigned short* HwT  = (unsigned short*)(ws + (1u << 20));
  unsigned short* Tbf  = (unsigned short*)((char*)HwT + (size_t)DOUT * NV * 2);
  unsigned short* M    = (unsigned short*)((char*)Tbf + (size_t)NV * NE * 2);

  cvt_T_kernel<<<((size_t)NV * NE) / (256 * 8), 256, 0, stream>>>(T, Tbf);
  gate_kernel<<<NE / 8, 256, 0, stream>>>(H_e, p, gate);
  hw_kernel<<<dim3(NV / 128, DOUT / 128), 256, 0, stream>>>(H_v, weight, HwT);
  passA_kernel<<<dim3(NV / 128, NV / 128), 256, 0, stream>>>(Tbf, gate, adj_v, M);
  passB_kernel<<<dim3(NV / 128, DOUT / 128), 256, 0, stream>>>(M, HwT, bias, out);

  // second tuple element: H_e passthrough
  (void)hipMemcpyAsync(out + (size_t)NV * DOUT, H_e,
                       sizeof(float) * (size_t)NE * DIN,
                       hipMemcpyDeviceToDevice, stream);
}
